// SharedChannelAttention_4509715661330
// MI455X (gfx1250) — matrix operand order repaired and checked by reading, not compiled, not run on hardware
//
#include <hip/hip_runtime.h>

#define EPS 1e-6f

typedef __bf16 bf16;
typedef __bf16 v16bf __attribute__((ext_vector_type(16)));
typedef __bf16 v8bf  __attribute__((ext_vector_type(8)));
typedef float  v8f   __attribute__((ext_vector_type(8)));
typedef float  v4f   __attribute__((ext_vector_type(4)));

static constexpr int B  = 8;
static constexpr int C  = 512;
static constexpr int NN = 16384;   // W*H
static constexpr int KC = 32;      // K chunk per WMMA step
static constexpr int PAD = 40;     // bf16 row stride: 80B = 16B aligned, bank-spread

// ---- Fragment builders (documented CDNA5 wave32 16-bit layouts) -------------
// A 16x32 bf16: lane l(0..15)/half h: elem e<8 -> K = h*8+e ; e>=8 -> K = 16+h*8+(e-8)
__device__ __forceinline__ v16bf ldsA_frag(const bf16* row, int h16) {
    v8bf lo = *(const v8bf*)(row + h16 * 8);
    v8bf hi = *(const v8bf*)(row + 16 + h16 * 8);
    v16bf a;
#pragma unroll
    for (int i = 0; i < 8; ++i) { a[i] = lo[i]; a[8 + i] = hi[i]; }
    return a;
}
// B 32x16 bf16: lane l = column N, elem e<8 -> K = h*8+e ; e>=8 -> K = 16+h*8+(e-8)
__device__ __forceinline__ v16bf ldsB_frag(const bf16* row, int h16) {
    v8bf lo = *(const v8bf*)(row + h16 * 8);
    v8bf hi = *(const v8bf*)(row + 16 + h16 * 8);
    v16bf b;
#pragma unroll
    for (int i = 0; i < 8; ++i) { b[i] = lo[i]; b[8 + i] = hi[i]; }
    return b;
}

// ---- Phase 1: per-(b,n) column stats ---------------------------------------
__global__ void __launch_bounds__(256)
colstats_kernel(const float* __restrict__ x, float* __restrict__ invn,
                float* __restrict__ vsum, float* __restrict__ wv) {
    int gid = blockIdx.x * 256 + threadIdx.x;     // b*NN + n
    int b = gid >> 14;
    int n = gid & (NN - 1);
    const float* p = x + (size_t)b * C * NN + n;
    float s = 0.f, ss = 0.f;
#pragma unroll 4
    for (int c = 0; c < C; ++c) {
        float v = p[(size_t)c * NN];
        s += v; ss += v * v;
    }
    float in = rsqrtf(ss);                        // 1/||column||
    invn[gid] = in;
    vsum[gid] = s;
    wv[gid]   = in * (s * in + EPS);              // inv_norm * q_colsum
}

// ---- Phase 2: matrix[b] = V * (V*diag(invn))^T  (C x C, K = NN) ------------
// Blocks with blockIdx.y==0 additionally fold in the tailor matvec:
//   tailor[b,c] = 1/(N + sum_n V[c,n] * w[n])   (f32, from the A-panel stream)
__global__ void __launch_bounds__(256)
gram_kernel(const float* __restrict__ x, const float* __restrict__ invn,
            const float* __restrict__ wv, float* __restrict__ matrix,
            float* __restrict__ tailor) {
    __shared__ __align__(16) bf16 Apan[128][PAD];
    __shared__ __align__(16) bf16 Bpan[128][PAD];
    __shared__ float tred[128][8];
    int b  = blockIdx.z;
    int m0 = blockIdx.x * 128;   // output rows (c)
    int n0 = blockIdx.y * 128;   // output cols (m)
    const float* xa  = x    + (size_t)b * C * NN;
    const float* inv = invn + (size_t)b * NN;
    const float* w   = wv   + (size_t)b * NN;
    const bool doTail = (blockIdx.y == 0);

    int tid = threadIdx.x, lane = tid & 31, wave = tid >> 5;
    int h16 = lane >> 4, l = lane & 15;
    int wr = wave >> 1, wc = wave & 1;

    v8f acc[2][4];
#pragma unroll
    for (int t = 0; t < 2; ++t)
#pragma unroll
        for (int u = 0; u < 4; ++u)
#pragma unroll
            for (int i = 0; i < 8; ++i) acc[t][u][i] = 0.f;

    float tacc[4] = {0.f, 0.f, 0.f, 0.f};
    int lr = tid >> 3;            // row 0..31 per stage step
    int lk = (tid & 7) * 4;       // k quad

    for (int kk = 0; kk < NN; kk += KC) {
        v4f iv4 = *(const v4f*)(inv + kk + lk);
        v4f wq4 = *(const v4f*)(w + kk + lk);
#pragma unroll
        for (int it = 0; it < 4; ++it) {
            int r = lr + it * 32;
            v4f va = *(const v4f*)(xa + (size_t)(m0 + r) * NN + kk + lk);
            v4f vb = *(const v4f*)(xa + (size_t)(n0 + r) * NN + kk + lk);
#pragma unroll
            for (int j = 0; j < 4; ++j) {
                Apan[r][lk + j] = (bf16)va[j];
                Bpan[r][lk + j] = (bf16)(vb[j] * iv4[j]);
            }
            if (doTail) {
#pragma unroll
                for (int j = 0; j < 4; ++j) tacc[it] += va[j] * wq4[j];
            }
        }
        if (kk + KC < NN) {   // pull next K chunk toward L2/L0
            __builtin_prefetch(xa + (size_t)(m0 + lr) * NN + kk + KC + lk, 0, 1);
            __builtin_prefetch(xa + (size_t)(n0 + lr) * NN + kk + KC + lk, 0, 1);
        }
        __syncthreads();

        v16bf afr[2], bfr[4];
#pragma unroll
        for (int t = 0; t < 2; ++t) afr[t] = ldsA_frag(&Apan[wr * 32 + t * 16 + l][0], h16);
#pragma unroll
        for (int u = 0; u < 4; ++u) bfr[u] = ldsB_frag(&Bpan[wc * 64 + u * 16 + l][0], h16);
#pragma unroll
        for (int t = 0; t < 2; ++t)
#pragma unroll
            for (int u = 0; u < 4; ++u)
                acc[t][u] = __builtin_amdgcn_wmma_f32_16x16x32_bf16(
                    false, afr[t], false, bfr[u], (short)0, acc[t][u], false, false);
        __syncthreads();
    }

    // tailor reduction (only blockIdx.y == 0 blocks; rows unique per block)
    if (doTail) {
#pragma unroll
        for (int it = 0; it < 4; ++it) tred[lr + it * 32][tid & 7] = tacc[it];
        __syncthreads();
        if (tid < 128) {
            float s = 0.f;
#pragma unroll
            for (int j = 0; j < 8; ++j) s += tred[tid][j];
            tailor[b * C + m0 + tid] = 1.0f / ((float)NN + s);
        }
    }

    float* mout = matrix + (size_t)b * C * C;
#pragma unroll
    for (int t = 0; t < 2; ++t)
#pragma unroll
        for (int u = 0; u < 4; ++u) {
            int nl = n0 + wc * 64 + u * 16 + l;
#pragma unroll
            for (int i = 0; i < 8; ++i) {
                int cl = m0 + wr * 32 + t * 16 + i + 8 * h16;
                mout[(size_t)cl * C + nl] = acc[t][u][i];
            }
        }
}

// ---- Phase 3: out = x + gamma * tailor[c] * (vsum[n] + matrix @ Qn) --------
__global__ void __launch_bounds__(256)
attn_out_kernel(const float* __restrict__ x, const float* __restrict__ matrix,
                const float* __restrict__ invn, const float* __restrict__ vsum,
                const float* __restrict__ tailor, const float* __restrict__ gamma,
                float* __restrict__ out) {
    __shared__ __align__(16) bf16 Apan[128][PAD];
    __shared__ __align__(16) bf16 Bpan[128][PAD];
    int b  = blockIdx.z;
    int m0 = blockIdx.x * 128;   // output rows (c)
    int n0 = blockIdx.y * 128;   // output cols (n)
    const float* xb  = x      + (size_t)b * C * NN;
    const float* ma  = matrix + (size_t)b * C * C;
    const float* inv = invn   + (size_t)b * NN;

    int tid = threadIdx.x, lane = tid & 31, wave = tid >> 5;
    int h16 = lane >> 4, l = lane & 15;
    int wr = wave >> 1, wc = wave & 1;

    v8f acc[2][4];
#pragma unroll
    for (int t = 0; t < 2; ++t)
#pragma unroll
        for (int u = 0; u < 4; ++u)
#pragma unroll
            for (int i = 0; i < 8; ++i) acc[t][u][i] = 0.f;

    int lr = tid >> 3;            // A: row 0..31 per step
    int lk = (tid & 7) * 4;       // A: k quad
    int bk = tid >> 5;            // B: k row 0..7 per step
    int bj = (tid & 31) * 4;      // B: n quad

    for (int kk = 0; kk < C; kk += KC) {
        // A: matrix[m0+r][kk+k], float4, coalesced per 8-thread group
#pragma unroll
        for (int it = 0; it < 4; ++it) {
            int r = lr + it * 32;
            v4f va = *(const v4f*)(ma + (size_t)(m0 + r) * C + kk + lk);
#pragma unroll
            for (int j = 0; j < 4; ++j) Apan[r][lk + j] = (bf16)va[j];
        }
        // B: Qn[kk+k][n0+j] = x * invn, float4 coalesced in n, transposed into [j][k]
        v4f iv4 = *(const v4f*)(inv + n0 + bj);
#pragma unroll
        for (int it = 0; it < 4; ++it) {
            int k = bk + it * 8;
            v4f vb = *(const v4f*)(xb + (size_t)(kk + k) * NN + n0 + bj);
#pragma unroll
            for (int j = 0; j < 4; ++j) Bpan[bj + j][k] = (bf16)(vb[j] * iv4[j]);
        }
        __syncthreads();

        v16bf afr[2], bfr[4];
#pragma unroll
        for (int t = 0; t < 2; ++t) afr[t] = ldsA_frag(&Apan[wr * 32 + t * 16 + l][0], h16);
#pragma unroll
        for (int u = 0; u < 4; ++u) bfr[u] = ldsB_frag(&Bpan[wc * 64 + u * 16 + l][0], h16);
#pragma unroll
        for (int t = 0; t < 2; ++t)
#pragma unroll
            for (int u = 0; u < 4; ++u)
                acc[t][u] = __builtin_amdgcn_wmma_f32_16x16x32_bf16(
                    false, afr[t], false, bfr[u], (short)0, acc[t][u], false, false);
        __syncthreads();
    }

    float g = gamma[0];
    const float* vs = vsum   + (size_t)b * NN;
    const float* tl = tailor + (size_t)b * C;
    float* ob = out + (size_t)b * C * NN;
#pragma unroll
    for (int t = 0; t < 2; ++t)
#pragma unroll
        for (int u = 0; u < 4; ++u) {
            int nl = n0 + wc * 64 + u * 16 + l;
            float vsn = vs[nl];
#pragma unroll
            for (int i = 0; i < 8; ++i) {
                int cl = m0 + wr * 32 + t * 16 + i + 8 * h16;
                size_t idx = (size_t)cl * NN + nl;
                ob[idx] = xb[idx] + g * tl[cl] * (vsn + acc[t][u][i]);
            }
        }
}

extern "C" void kernel_launch(void* const* d_in, const int* in_sizes, int n_in,
                              void* d_out, int out_size, void* d_ws, size_t ws_size,
                              hipStream_t stream) {
    (void)in_sizes; (void)n_in; (void)out_size; (void)ws_size;
    const float* x     = (const float*)d_in[0];
    const float* gamma = (const float*)d_in[1];
    float* out = (float*)d_out;

    float* ws   = (float*)d_ws;
    float* invn = ws;                       // B*NN = 131072 f32
    float* vsum = ws + 131072;              // B*NN
    float* wv   = ws + 262144;              // B*NN
    float* tl   = ws + 393216;              // B*C  = 4096
    float* mat  = ws + 397312;              // B*C*C = 2097152  (~10 MB total)

    colstats_kernel<<<(B * NN) / 256, 256, 0, stream>>>(x, invn, vsum, wv);
    gram_kernel<<<dim3(4, 4, B), 256, 0, stream>>>(x, invn, wv, mat, tl);
    attn_out_kernel<<<dim3(4, 128, B), 256, 0, stream>>>(x, mat, invn, vsum, tl, gamma, out);
}
